// RNN_GAUSS_37039797961350
// MI455X (gfx1250) — matrix operand order repaired and checked by reading, not compiled, not run on hardware
//
#include <hip/hip_runtime.h>

#define DEVINL __device__ __forceinline__

typedef __attribute__((ext_vector_type(16))) __bf16 v16bf;
typedef __attribute__((ext_vector_type(8)))  float  v8f;

static constexpr int Tt = 256, Bb = 1024, Yy = 32, Hh = 1024, Rr = 512;
static constexpr float LOG2PI = 1.83787706640934548356f;
static constexpr int NLL_BLOCKS = 8;

struct U4x2 { uint4 lo, hi; };

DEVINL v16bf mk_frag(uint4 lo, uint4 hi) {
  U4x2 t; t.lo = lo; t.hi = hi;
  return __builtin_bit_cast(v16bf, t);
}

DEVINL __bf16 f2bf(float f) {  // round-to-nearest-even fp32 -> bf16
  unsigned u = __builtin_bit_cast(unsigned, f);
  u = (u + 0x7FFFu + ((u >> 16) & 1u)) >> 16;
  unsigned short s = (unsigned short)u;
  return __builtin_bit_cast(__bf16, s);
}

// ---------------------------------------------------------------------------
// fp32 -> bf16 conversion (weights / y), done once per launch
// ---------------------------------------------------------------------------
__global__ __launch_bounds__(256) void cvt_bf16_kernel(const float* __restrict__ src,
                                                       __bf16* __restrict__ dst, int n) {
  int i = blockIdx.x * 256 + threadIdx.x;
  if (i < n) dst[i] = f2bf(src[i]);
}

// ---------------------------------------------------------------------------
// Tiled WMMA GEMM:  Out[M,N] = act(A[M,K] @ W[N,K]^T + bias[N])
//   bf16 in, fp32 accumulate (v_wmma_f32_16x16x32_bf16).
//   Block: 256 threads = 8 wave32, 128x64 tile, BK=32, double-buffered LDS,
//   ONE barrier per K-chunk. Wave (wm 0..3, wn 0..1) owns a 32x32 sub-tile:
//   2 A frags x 2 B frags -> 4 accumulators, 4 WMMAs per chunk.
//   M assumed multiple of 128 (M == 1024 here). N may be ragged (N == 32).
// ---------------------------------------------------------------------------
template <bool RELU_BF16>
__global__ __launch_bounds__(256) void gemm_wmma_kernel(
    const __bf16* __restrict__ A, const __bf16* __restrict__ W,
    const float* __restrict__ bias, void* __restrict__ Out,
    int M, int N, int K) {
  __shared__ alignas(16) __bf16 As[2][128 * 32];  // 16 KB
  __shared__ alignas(16) __bf16 Bs[2][64 * 32];   //  8 KB

  const int tid  = threadIdx.x;
  const int lane = tid & 31;
  const int l16  = lane & 15;
  const int lhi  = lane >> 4;          // half-wave select
  const int wave = tid >> 5;
  const int wm   = wave & 3;           // row group: wm*32
  const int wn   = wave >> 2;          // col group: wn*32

  const int m_blk = blockIdx.y * 128;
  const int n_blk = blockIdx.x * 64;

  const int ldr = tid >> 2;            // 0..63 : staging row
  const int seg = (tid & 3) * 8;       // 16-byte segment within 32-elem row

  const size_t a_row0 = (size_t)(m_blk + ldr) * K;        // A rows ldr, ldr+64
  const size_t a_row1 = (size_t)(m_blk + ldr + 64) * K;
  const bool   b_ok   = (n_blk + ldr) < N;
  const size_t b_row  = (size_t)(n_blk + ldr) * K;

  v8f acc00 = {}, acc01 = {}, acc10 = {}, acc11 = {};

  const int nch = K >> 5;  // K is a multiple of 32 (512 / 1024 / 32)

  // ---- prologue: stage chunk 0 ----
  {
    uint4 a0 = *reinterpret_cast<const uint4*>(A + a_row0 + seg);
    uint4 a1 = *reinterpret_cast<const uint4*>(A + a_row1 + seg);
    uint4 b0 = make_uint4(0u, 0u, 0u, 0u);
    if (b_ok) b0 = *reinterpret_cast<const uint4*>(W + b_row + seg);
    *reinterpret_cast<uint4*>(&As[0][ldr * 32 + seg])        = a0;
    *reinterpret_cast<uint4*>(&As[0][(ldr + 64) * 32 + seg]) = a1;
    *reinterpret_cast<uint4*>(&Bs[0][ldr * 32 + seg])        = b0;
  }
  __syncthreads();

  for (int i = 0; i < nch; ++i) {
    const int cur = i & 1, nxt = cur ^ 1;
    const bool more = (i + 1) < nch;

    // issue global loads for chunk i+1 early (compiler overlaps with math)
    uint4 na0, na1, nb0;
    if (more) {
      const int kc = (i + 1) << 5;
      na0 = *reinterpret_cast<const uint4*>(A + a_row0 + kc + seg);
      na1 = *reinterpret_cast<const uint4*>(A + a_row1 + kc + seg);
      nb0 = make_uint4(0u, 0u, 0u, 0u);
      if (b_ok) nb0 = *reinterpret_cast<const uint4*>(W + b_row + kc + seg);
    }

    // ---- fragments from LDS buffer `cur` ----
    const __bf16* ap0 = &As[cur][(wm * 32 + l16) * 32 + lhi * 8];
    const __bf16* ap1 = ap0 + 16 * 32;
    v16bf a0 = mk_frag(*reinterpret_cast<const uint4*>(ap0),
                       *reinterpret_cast<const uint4*>(ap0 + 16));
    v16bf a1 = mk_frag(*reinterpret_cast<const uint4*>(ap1),
                       *reinterpret_cast<const uint4*>(ap1 + 16));

    const __bf16* bp0 = &Bs[cur][(wn * 32 + l16) * 32 + lhi * 8];
    const __bf16* bp1 = bp0 + 16 * 32;
    v16bf b0 = mk_frag(*reinterpret_cast<const uint4*>(bp0),
                       *reinterpret_cast<const uint4*>(bp0 + 16));
    v16bf b1 = mk_frag(*reinterpret_cast<const uint4*>(bp1),
                       *reinterpret_cast<const uint4*>(bp1 + 16));

    acc00 = __builtin_amdgcn_wmma_f32_16x16x32_bf16(false, a0, false, b0, (short)0, acc00, false, false);
    acc01 = __builtin_amdgcn_wmma_f32_16x16x32_bf16(false, a0, false, b1, (short)0, acc01, false, false);
    acc10 = __builtin_amdgcn_wmma_f32_16x16x32_bf16(false, a1, false, b0, (short)0, acc10, false, false);
    acc11 = __builtin_amdgcn_wmma_f32_16x16x32_bf16(false, a1, false, b1, (short)0, acc11, false, false);

    // stage chunk i+1 into the other buffer (safe: previous reads of `nxt`
    // were fenced by the barrier at the end of iteration i-1)
    if (more) {
      *reinterpret_cast<uint4*>(&As[nxt][ldr * 32 + seg])        = na0;
      *reinterpret_cast<uint4*>(&As[nxt][(ldr + 64) * 32 + seg]) = na1;
      *reinterpret_cast<uint4*>(&Bs[nxt][ldr * 32 + seg])        = nb0;
    }
    __syncthreads();
  }

  // ---- epilogue: lane l16 = col, VGPR v + lhi*8 = row within 16x16 tile ----
  const int col0 = n_blk + wn * 32 + l16;
  const int col1 = col0 + 16;
  const int rbase = m_blk + wm * 32 + lhi * 8;
  const float bia0 = (col0 < N) ? bias[col0] : 0.f;
  const float bia1 = (col1 < N) ? bias[col1] : 0.f;

  v8f accs0[2] = {acc00, acc10};  // col0 accumulators, ai = 0,1
  v8f accs1[2] = {acc01, acc11};  // col1 accumulators
#pragma unroll
  for (int ai = 0; ai < 2; ++ai) {
#pragma unroll
    for (int v = 0; v < 8; ++v) {
      const int r = rbase + ai * 16 + v;
      float x0 = accs0[ai][v] + bia0;
      float x1 = accs1[ai][v] + bia1;
      if (RELU_BF16) {
        x0 = fmaxf(x0, 0.f);
        x1 = fmaxf(x1, 0.f);
        __bf16* o = reinterpret_cast<__bf16*>(Out);
        if (col0 < N) o[(size_t)r * N + col0] = f2bf(x0);
        if (col1 < N) o[(size_t)r * N + col1] = f2bf(x1);
      } else {
        float* o = reinterpret_cast<float*>(Out);
        if (col0 < N) o[(size_t)r * N + col0] = x0;
        if (col1 < N) o[(size_t)r * N + col1] = x1;
      }
    }
  }
}

// ---------------------------------------------------------------------------
// GRU elementwise combine (torch gate order r,z,n); updates h fp32 + bf16
// ---------------------------------------------------------------------------
DEVINL float sigmoidf_(float x) { return 1.f / (1.f + __expf(-x)); }

__global__ __launch_bounds__(256) void gru_combine_kernel(
    const float* __restrict__ gi, const float* __restrict__ gh,
    float* __restrict__ h_f, __bf16* __restrict__ h_b, int total, int R) {
  int idx = blockIdx.x * 256 + threadIdx.x;
  if (idx >= total) return;
  int b = idx / R, j = idx - b * R;
  const float* gib = gi + (size_t)b * 3 * R;
  const float* ghb = gh + (size_t)b * 3 * R;
  float r = sigmoidf_(gib[j] + ghb[j]);
  float z = sigmoidf_(gib[j + R] + ghb[j + R]);
  float n = tanhf(gib[j + 2 * R] + r * ghb[j + 2 * R]);
  float h = h_f[idx];
  float hn = (1.f - z) * n + z * h;
  h_f[idx] = hn;
  h_b[idx] = f2bf(hn);
}

// ---------------------------------------------------------------------------
// Per-step Gaussian NLL: NLL_BLOCKS deterministic partial reductions
// ---------------------------------------------------------------------------
__global__ __launch_bounds__(1024) void nll_step_kernel(
    const float* __restrict__ y_t, const float* __restrict__ mean,
    const float* __restrict__ stdp, float* __restrict__ out_part) {
  __shared__ float buf[1024];
  const int chunk = (Bb * Yy) / NLL_BLOCKS;          // 4096
  const int base = blockIdx.x * chunk;
  float s = 0.f;
  for (int i = threadIdx.x; i < chunk; i += 1024) {
    int g = base + i;
    float x = stdp[g];
    float st = (x > 20.f) ? x : log1pf(__expf(x));   // softplus
    float d = y_t[g] - mean[g];
    s += d * d / (st * st) + 2.f * logf(st) + LOG2PI;
  }
  buf[threadIdx.x] = s;
  __syncthreads();
  for (int o = 512; o > 0; o >>= 1) {
    if (threadIdx.x < (unsigned)o) buf[threadIdx.x] += buf[threadIdx.x + o];
    __syncthreads();
  }
  if (threadIdx.x == 0) out_part[blockIdx.x] = 0.5f * buf[0];
}

__global__ __launch_bounds__(256) void nll_final_kernel(const float* __restrict__ parts,
                                                        float* __restrict__ out) {
  __shared__ float buf[256];
  float s = 0.f;
  const int n = Tt * NLL_BLOCKS;                     // 2048
  for (int i = threadIdx.x; i < n; i += 256) s += parts[i];
  buf[threadIdx.x] = s;
  __syncthreads();
  for (int o = 128; o > 0; o >>= 1) {
    if (threadIdx.x < (unsigned)o) buf[threadIdx.x] += buf[threadIdx.x + o];
    __syncthreads();
  }
  if (threadIdx.x == 0) out[0] = buf[0];
}

// ---------------------------------------------------------------------------
// host side
// ---------------------------------------------------------------------------
static inline int cdiv(int a, int b) { return (a + b - 1) / b; }

static void cvt(const float* s, __bf16* d, int n, hipStream_t st) {
  cvt_bf16_kernel<<<cdiv(n, 256), 256, 0, st>>>(s, d, n);
}
static void gemm_relu_bf(const __bf16* A, const __bf16* W, const float* bias,
                         __bf16* out, int M, int N, int K, hipStream_t st) {
  dim3 g(cdiv(N, 64), cdiv(M, 128));
  gemm_wmma_kernel<true><<<g, 256, 0, st>>>(A, W, bias, (void*)out, M, N, K);
}
static void gemm_f32(const __bf16* A, const __bf16* W, const float* bias,
                     float* out, int M, int N, int K, hipStream_t st) {
  dim3 g(cdiv(N, 64), cdiv(M, 128));
  gemm_wmma_kernel<false><<<g, 256, 0, st>>>(A, W, bias, (void*)out, M, N, K);
}

extern "C" void kernel_launch(void* const* d_in, const int* in_sizes, int n_in,
                              void* d_out, int out_size, void* d_ws, size_t ws_size,
                              hipStream_t stream) {
  (void)in_sizes; (void)n_in; (void)out_size; (void)ws_size;

  const float* y       = (const float*)d_in[0];
  const float* dec_w1  = (const float*)d_in[1];
  const float* dec_b1  = (const float*)d_in[2];
  const float* dec_w2  = (const float*)d_in[3];
  const float* dec_b2  = (const float*)d_in[4];
  const float* mean_w  = (const float*)d_in[5];
  const float* mean_b  = (const float*)d_in[6];
  const float* std_w   = (const float*)d_in[7];
  const float* std_b   = (const float*)d_in[8];
  const float* wih0    = (const float*)d_in[9];
  const float* whh0    = (const float*)d_in[10];
  const float* bih0    = (const float*)d_in[11];
  const float* bhh0    = (const float*)d_in[12];
  const float* wih1    = (const float*)d_in[13];
  const float* whh1    = (const float*)d_in[14];
  const float* bih1    = (const float*)d_in[15];
  const float* bhh1    = (const float*)d_in[16];

  char* ws = (char*)d_ws;
  size_t off = 0;
  auto alloc = [&](size_t bytes) -> char* {
    char* p = ws + off;
    off = (off + bytes + 255) & ~size_t(255);
    return p;
  };

  __bf16* wb_dec1 = (__bf16*)alloc((size_t)Hh * Rr * 2);
  __bf16* wb_dec2 = (__bf16*)alloc((size_t)Hh * Hh * 2);
  __bf16* wb_mean = (__bf16*)alloc((size_t)Yy * Hh * 2);
  __bf16* wb_std  = (__bf16*)alloc((size_t)Yy * Hh * 2);
  __bf16* wb_ih0  = (__bf16*)alloc((size_t)3 * Rr * Yy * 2);
  __bf16* wb_hh0  = (__bf16*)alloc((size_t)3 * Rr * Rr * 2);
  __bf16* wb_ih1  = (__bf16*)alloc((size_t)3 * Rr * Rr * 2);
  __bf16* wb_hh1  = (__bf16*)alloc((size_t)3 * Rr * Rr * 2);
  __bf16* y_bf    = (__bf16*)alloc((size_t)Tt * Bb * Yy * 2);
  float*  h0_f    = (float*) alloc((size_t)Bb * Rr * 4);
  float*  h1_f    = (float*) alloc((size_t)Bb * Rr * 4);
  __bf16* h0_b    = (__bf16*)alloc((size_t)Bb * Rr * 2);
  __bf16* h1_b    = (__bf16*)alloc((size_t)Bb * Rr * 2);
  __bf16* d1_b    = (__bf16*)alloc((size_t)Bb * Hh * 2);
  __bf16* d2_b    = (__bf16*)alloc((size_t)Bb * Hh * 2);
  float*  gif     = (float*) alloc((size_t)Bb * 3 * Rr * 4);
  float*  ghf     = (float*) alloc((size_t)Bb * 3 * Rr * 4);
  float*  meanf   = (float*) alloc((size_t)Bb * Yy * 4);
  float*  stdpf   = (float*) alloc((size_t)Bb * Yy * 4);
  float*  nll_pt  = (float*) alloc((size_t)Tt * NLL_BLOCKS * 4);

  // one-time (per call) conversions: weights + y to bf16
  cvt(dec_w1, wb_dec1, Hh * Rr, stream);
  cvt(dec_w2, wb_dec2, Hh * Hh, stream);
  cvt(mean_w, wb_mean, Yy * Hh, stream);
  cvt(std_w,  wb_std,  Yy * Hh, stream);
  cvt(wih0,   wb_ih0,  3 * Rr * Yy, stream);
  cvt(whh0,   wb_hh0,  3 * Rr * Rr, stream);
  cvt(wih1,   wb_ih1,  3 * Rr * Rr, stream);
  cvt(whh1,   wb_hh1,  3 * Rr * Rr, stream);
  cvt(y,      y_bf,    Tt * Bb * Yy, stream);

  // h starts at zero each call
  hipMemsetAsync(h0_f, 0, (size_t)Bb * Rr * 4, stream);
  hipMemsetAsync(h1_f, 0, (size_t)Bb * Rr * 4, stream);
  hipMemsetAsync(h0_b, 0, (size_t)Bb * Rr * 2, stream);
  hipMemsetAsync(h1_b, 0, (size_t)Bb * Rr * 2, stream);

  const int combine_blocks = cdiv(Bb * Rr, 256);

  for (int t = 0; t < Tt; ++t) {
    const float*  y_t    = y    + (size_t)t * Bb * Yy;
    const __bf16* y_t_bf = y_bf + (size_t)t * Bb * Yy;

    // decoder (uses pre-update top hidden state h1)
    gemm_relu_bf(h1_b, wb_dec1, dec_b1, d1_b, Bb, Hh, Rr, stream);
    gemm_relu_bf(d1_b, wb_dec2, dec_b2, d2_b, Bb, Hh, Hh, stream);
    gemm_f32(d2_b, wb_mean, mean_b, meanf, Bb, Yy, Hh, stream);
    gemm_f32(d2_b, wb_std,  std_b,  stdpf, Bb, Yy, Hh, stream);
    nll_step_kernel<<<NLL_BLOCKS, 1024, 0, stream>>>(y_t, meanf, stdpf,
                                                     nll_pt + (size_t)t * NLL_BLOCKS);

    // GRU layer 0
    gemm_f32(y_t_bf, wb_ih0, bih0, gif, Bb, 3 * Rr, Yy, stream);
    gemm_f32(h0_b,   wb_hh0, bhh0, ghf, Bb, 3 * Rr, Rr, stream);
    gru_combine_kernel<<<combine_blocks, 256, 0, stream>>>(gif, ghf, h0_f, h0_b,
                                                           Bb * Rr, Rr);
    // GRU layer 1 (gi from NEW h0, gh from OLD h1)
    gemm_f32(h0_b, wb_ih1, bih1, gif, Bb, 3 * Rr, Rr, stream);
    gemm_f32(h1_b, wb_hh1, bhh1, ghf, Bb, 3 * Rr, Rr, stream);
    gru_combine_kernel<<<combine_blocks, 256, 0, stream>>>(gif, ghf, h1_f, h1_b,
                                                           Bb * Rr, Rr);
  }

  nll_final_kernel<<<1, 256, 0, stream>>>(nll_pt, (float*)d_out);
}
